// ComprehensionModel_64158221468310
// MI455X (gfx1250) — matrix operand order repaired and checked by reading, not compiled, not run on hardware
//
#include <hip/hip_runtime.h>
#include <math.h>

typedef __attribute__((ext_vector_type(16))) _Float16 v16h;
typedef __attribute__((ext_vector_type(8)))  float    v8f;
typedef __attribute__((ext_vector_type(4)))  unsigned int v4u;
typedef __attribute__((ext_vector_type(8)))  int      v8i;
typedef __attribute__((ext_vector_type(4)))  int      v4i;

#define B_ 64
#define T_ 512
#define V_ 512
#define H_ 256
#define O_ 512

// ---- WMMA tile loaders (wave32, contiguous k order in A and B, CDNA5 ISA §7.12.2) ----

// A-matrix 16x32 f16 (MxK): lane L holds row M=L%16; lanes<16: K=0..15, lanes>=16: K=16..31.
__device__ __forceinline__ v16h load_A16(const _Float16* __restrict__ p0, int ld, int lane) {
  int half = lane >> 4;
  int m    = lane & 15;
  const _Float16* p = p0 + (size_t)m * ld + half * 16;
  v16h a;
#pragma unroll
  for (int j = 0; j < 8; ++j) a[j] = p[j];          // K = half*16 + j
#pragma unroll
  for (int j = 0; j < 8; ++j) a[8 + j] = p[8 + j];  // K = half*16 + 8 + j
  return a;
}

// B-matrix 32x16 f16 (KxN) from row-major W[N][K] (B[k][n] = W[n][k]):
// lane L holds column N=L%16; lanes 0-15: K=0..15, lanes 16-31: K=16..31 (32 contiguous bytes).
__device__ __forceinline__ v16h load_BT16(const _Float16* __restrict__ w0, int ld, int lane) {
  int half = lane >> 4;
  int n    = lane & 15;
  const _Float16* p = w0 + (size_t)n * ld + half * 16;
  v16h b;
#pragma unroll
  for (int j = 0; j < 16; ++j) b[j] = p[j];
  return b;
}

// ---- Tensor Data Mover: 1-D f16 slab (B_*H_ elements) global -> LDS (CDNA5 ISA) ----
__device__ __forceinline__ void tdm_load_xp(const _Float16* gsrc, unsigned int lds_byte_off) {
  unsigned long long ga = (unsigned long long)(size_t)gsrc;
  const unsigned int NE = (unsigned int)(B_ * H_);       // 16384 elements
  v4u g0;
  g0[0] = 1u;                                            // count=1, user descriptor
  g0[1] = lds_byte_off;                                  // lds_addr
  g0[2] = (unsigned int)ga;                              // global_addr[31:0]
  g0[3] = (unsigned int)((ga >> 32) & 0x1FFFFFFu) | (2u << 30); // addr[56:32] | type=2
  v8i g1;
  g1[0] = (int)(1u << 16);                               // wg_mask=0, data_size=1 (2B)
  g1[1] = (int)((NE & 0xFFFFu) << 16);                   // tensor_dim0[15:0]
  g1[2] = (int)((NE >> 16) | (1u << 16));                // tensor_dim0[31:16] | tensor_dim1=1
  g1[3] = (int)((NE & 0xFFFFu) << 16);                   // tile_dim0 = NE
  g1[4] = 1;                                             // tile_dim1 = 1
  g1[5] = (int)NE;                                       // tensor_dim0_stride
  g1[6] = 0;
  g1[7] = 0;
  v4i g2 = {0, 0, 0, 0};
  v4i g3 = {0, 0, 0, 0};
  v8i g4 = {0, 0, 0, 0, 0, 0, 0, 0};                     // 6-arg toolchain form
  __builtin_amdgcn_tensor_load_to_lds(g0, g1, g2, g3, g4, 0);
}

// ---- elementwise f32 -> f16 conversion ----
__global__ __launch_bounds__(256) void cvt_f32_f16(const float* __restrict__ in,
                                                   _Float16* __restrict__ out, int n) {
  int stride = gridDim.x * blockDim.x;
  for (int i = blockIdx.x * blockDim.x + threadIdx.x; i < n; i += stride)
    out[i] = (_Float16)in[i];
}

// ---- Phase 1: x_proj[t][b][h] = sentence[b,t,:] @ W_ih^T + b_ih  (time-major f16) ----
// Each wave: one M-tile x four N-tiles (A loaded once per K-step, 4 WMMAs).
__global__ __launch_bounds__(256) void xproj_gemm(const _Float16* __restrict__ sent,  // [B*T, V]
                                                  const _Float16* __restrict__ wih,   // [H, V]
                                                  const float* __restrict__ bih,
                                                  _Float16* __restrict__ xproj) {     // [T, B, H]
  int gw   = (blockIdx.x * blockDim.x + threadIdx.x) >> 5;
  int lane = threadIdx.x & 31;
  int tg   = gw & 3;                        // 4 groups of 4 N-tiles (H=256)
  int tm   = gw >> 2;                       // 2048 tiles along B*T
  size_t m0 = (size_t)tm * 16;
  int    n0 = tg * 64;
  const _Float16* aBase = sent + m0 * V_;
  v8f acc[4] = {};
  for (int k0 = 0; k0 < V_; k0 += 32) {
    v16h a = load_A16(aBase + k0, V_, lane);
#pragma unroll
    for (int i = 0; i < 4; ++i) {
      v16h b = load_BT16(wih + (size_t)(n0 + i * 16) * V_ + k0, V_, lane);
      acc[i] = __builtin_amdgcn_wmma_f32_16x16x32_f16(false, a, false, b, (short)0, acc[i],
                                                      false, false);
    }
  }
  int half = lane >> 4;
  int bidx = (int)(m0 / T_);                // 16 | T so b is tile-constant
  int t0   = (int)(m0 % T_);
#pragma unroll
  for (int i = 0; i < 4; ++i) {
    int n = n0 + i * 16 + (lane & 15);
    float bias = bih[n];
#pragma unroll
    for (int j = 0; j < 8; ++j) {
      int t = t0 + j + half * 8;            // C layout: VGPR j -> M = j + half*8
      xproj[((size_t)t * B_ + bidx) * H_ + n] = (_Float16)(acc[i][j] + bias);
    }
  }
}

// ---- Phase 2: serial scan. One WGP: W_hh 128KB + h 2x32KB + x_proj slabs 2x32KB in LDS.
//      TDM double-buffers the next step's x_proj slab while WMMAs run. ----
__global__ __launch_bounds__(1024) void rnn_scan(const float* __restrict__ prev_state, // [B,H]
                                                 const _Float16* __restrict__ whh,     // [H,H] f16
                                                 const float* __restrict__ bhh,        // [H]
                                                 const _Float16* __restrict__ xproj,   // [T,B,H] f16
                                                 _Float16* __restrict__ activ,         // [B*T, H] f16
                                                 float* __restrict__ hN) {             // [B,H]
  extern __shared__ char smem[];
  _Float16* s_whh = (_Float16*)smem;                        // 131072 B
  _Float16* s_h0  = (_Float16*)(smem + 131072);             //  32768 B
  _Float16* s_h1  = (_Float16*)(smem + 163840);             //  32768 B
  _Float16* s_xp0 = (_Float16*)(smem + 196608);             //  32768 B
  _Float16* s_xp1 = (_Float16*)(smem + 229376);             //  32768 B
  const unsigned int XP_OFF = 196608u;

  int tid = threadIdx.x;
  int wave = tid >> 5, lane = tid & 31;
  int half = lane >> 4, nl = lane & 15;

  if (wave == 0) {                                          // prefetch slab for t=0
    tdm_load_xp(xproj, XP_OFF);
  }
  for (int i = tid; i < H_ * H_; i += 1024) s_whh[i] = whh[i];
  for (int i = tid; i < B_ * H_; i += 1024) s_h0[i] = (_Float16)prev_state[i];
  if (wave == 0) __builtin_amdgcn_s_wait_tensorcnt(0);
  __syncthreads();

  for (int t = 0; t < T_; ++t) {
    if (wave == 0 && t + 1 < T_)                            // async prefetch slab t+1
      tdm_load_xp(xproj + (size_t)(t + 1) * B_ * H_, XP_OFF + (unsigned)((t + 1) & 1) * 32768u);

    _Float16* hc = (t & 1) ? s_h1 : s_h0;                   // read current state
    _Float16* hw = (t & 1) ? s_h0 : s_h1;                   // write next state
    const _Float16* xp = (t & 1) ? s_xp1 : s_xp0;
#pragma unroll
    for (int i = 0; i < 2; ++i) {                           // 32 waves x 2 tiles = 64 C-tiles
      int tile = wave * 2 + i;
      int b0 = (tile >> 4) * 16;                            // batch rows
      int n0 = (tile & 15) * 16;                            // hidden cols
      v8f acc = {};
#pragma unroll
      for (int k0 = 0; k0 < H_; k0 += 32) {
        v16h a = load_A16(hc + (size_t)b0 * H_ + k0, H_, lane);
        v16h b = load_BT16(s_whh + (size_t)n0 * H_ + k0, H_, lane);
        acc = __builtin_amdgcn_wmma_f32_16x16x32_f16(false, a, false, b, (short)0, acc,
                                                     false, false);
      }
      int n = n0 + nl;
      float bias = bhh[n];
#pragma unroll
      for (int j = 0; j < 8; ++j) {
        int brow = b0 + j + half * 8;
        float v = acc[j] + (float)xp[(size_t)brow * H_ + n] + bias;
        v = fmaxf(v, 0.0f);                                 // relu -> h_t
        hw[(size_t)brow * H_ + n] = (_Float16)v;
        // capped_lrelu(h) == min(h,1) since h >= 0
        activ[((size_t)brow * T_ + t) * H_ + n] = (_Float16)fminf(v, 1.0f);
        if (t == T_ - 1) hN[(size_t)brow * H_ + n] = v;     // h_n output
      }
    }
    if (wave == 0) __builtin_amdgcn_s_wait_tensorcnt(0);    // slab t+1 landed
    __syncthreads();
  }
}

// ---- Phase 3: tags = sigmoid(activ @ W_out^T + b_out); 4 N-tiles per wave, NT stores ----
__global__ __launch_bounds__(256) void out_gemm(const _Float16* __restrict__ activ, // [B*T, H]
                                                const _Float16* __restrict__ wout,  // [O, H]
                                                const float* __restrict__ bout,
                                                float* __restrict__ tags) {         // [B*T, O]
  int gw   = (blockIdx.x * blockDim.x + threadIdx.x) >> 5;
  int lane = threadIdx.x & 31;
  int tg   = gw & 7;                        // 8 groups of 4 N-tiles (O=512)
  size_t m0 = (size_t)(gw >> 3) * 16;       // 2048 tiles along B*T
  int    n0 = tg * 64;
  const _Float16* aBase = activ + m0 * H_;
  v8f acc[4] = {};
#pragma unroll
  for (int k0 = 0; k0 < H_; k0 += 32) {
    v16h a = load_A16(aBase + k0, H_, lane);
#pragma unroll
    for (int i = 0; i < 4; ++i) {
      v16h b = load_BT16(wout + (size_t)(n0 + i * 16) * H_ + k0, H_, lane);
      acc[i] = __builtin_amdgcn_wmma_f32_16x16x32_f16(false, a, false, b, (short)0, acc[i],
                                                      false, false);
    }
  }
  int half = lane >> 4;
#pragma unroll
  for (int i = 0; i < 4; ++i) {
    int n = n0 + i * 16 + (lane & 15);
    float bias = bout[n];
#pragma unroll
    for (int j = 0; j < 8; ++j) {
      size_t row = m0 + j + half * 8;
      float x = acc[i][j] + bias;
      float s = 1.0f / (1.0f + __expf(-x));
      __builtin_nontemporal_store(s, &tags[row * O_ + n]);  // streamed output, keep L2 for activ
    }
  }
}

extern "C" void kernel_launch(void* const* d_in, const int* in_sizes, int n_in,
                              void* d_out, int out_size, void* d_ws, size_t ws_size,
                              hipStream_t stream) {
  const float* sent = (const float*)d_in[0];   // [B,T,V]
  const float* prev = (const float*)d_in[1];   // [1,B,H]
  const float* wih  = (const float*)d_in[2];   // [H,V]
  const float* bih  = (const float*)d_in[3];   // [H]
  const float* whh  = (const float*)d_in[4];   // [H,H]
  const float* bhh  = (const float*)d_in[5];   // [H]
  const float* wout = (const float*)d_in[6];   // [O,H]
  const float* bout = (const float*)d_in[7];   // [O]

  float* tags = (float*)d_out;                           // [B,T,O]
  float* hN   = tags + (size_t)B_ * T_ * O_;             // [1,B,H] appended

  // Workspace layout (~67.6 MB)
  char* ws = (char*)d_ws;
  size_t off = 0;
  _Float16* sent_h  = (_Float16*)(ws + off); off += (size_t)B_ * T_ * V_ * 2;
  _Float16* wih_h   = (_Float16*)(ws + off); off += (size_t)H_ * V_ * 2;
  _Float16* whh_h   = (_Float16*)(ws + off); off += (size_t)H_ * H_ * 2;
  _Float16* wout_h  = (_Float16*)(ws + off); off += (size_t)O_ * H_ * 2;
  _Float16* xproj_h = (_Float16*)(ws + off); off += (size_t)T_ * B_ * H_ * 2;
  _Float16* activ   = (_Float16*)(ws + off);

  // f32 -> f16 staging
  cvt_f32_f16<<<4096, 256, 0, stream>>>(sent, sent_h, B_ * T_ * V_);
  cvt_f32_f16<<<256,  256, 0, stream>>>(wih,  wih_h,  H_ * V_);
  cvt_f32_f16<<<128,  256, 0, stream>>>(whh,  whh_h,  H_ * H_);
  cvt_f32_f16<<<256,  256, 0, stream>>>(wout, wout_h, O_ * H_);

  // Phase 1: 2048 M-tiles x 4 wave-groups, 8 waves/block -> 1024 blocks
  xproj_gemm<<<(B_ * T_ / 16) * 4 / 8, 256, 0, stream>>>(sent_h, wih_h, bih, xproj_h);

  // Phase 2: one workgroup, 262144 B dynamic LDS
  rnn_scan<<<1, 1024, 262144, stream>>>(prev, whh_h, bhh, xproj_h, activ, hN);

  // Phase 3: 2048 M-tiles x 8 wave-groups, 8 waves/block -> 2048 blocks
  out_gemm<<<(B_ * T_ / 16) * 8 / 8, 256, 0, stream>>>(activ, wout_h, bout, tags);
}
